// XPoolAggregator_62380105007671
// MI455X (gfx1250) — matrix operand order repaired and checked by reading, not compiled, not run on hardware
//
#include <hip/hip_runtime.h>

// ---------------------------------------------------------------------------
// XPoolAggregator for MI455X (gfx1250): bf16 WMMA GEMM pipeline, wave32.
// All matrix math runs on v_wmma_f32_16x16x32_bf16 with f32 accumulation.
// All WMMA fragments are fed from LDS : A fragments as two contiguous 16B runs,
// B fragments as strided per-element reads.
// ---------------------------------------------------------------------------

typedef __attribute__((ext_vector_type(16))) __bf16 v16bf;
typedef __attribute__((ext_vector_type(8)))  float  v8f;
typedef unsigned short u16;

#define D 512
#define HEADS 8
#define HD 64
#define NEG_INF -1e30f

// ---- bf16 helpers ---------------------------------------------------------
__device__ __forceinline__ u16 f2bf(float f) {
  unsigned u = __float_as_uint(f);
  unsigned r = u + 0x7FFFu + ((u >> 16) & 1u);   // round-to-nearest-even
  return (u16)(r >> 16);
}
__device__ __forceinline__ float bf2f(u16 s) {
  return __uint_as_float(((unsigned)s) << 16);
}
__device__ __forceinline__ __bf16 us2bf(u16 s) {
  union { u16 u; __bf16 b; } x; x.u = s; return x.b;
}

__device__ __forceinline__ v8f wmma_bf16(v16bf a, v16bf b, v8f c) {
  return __builtin_amdgcn_wmma_f32_16x16x32_bf16(
      /*neg_a=*/false, a, /*neg_b=*/false, b,
      /*c_mod=*/(short)0, c, /*reuse_a=*/false, /*reuse_b=*/false);
}

// ---------------------------------------------------------------------------
// Kernel: f32 -> bf16 conversion (weights)
// ---------------------------------------------------------------------------
__global__ void f2bf_kernel(const float* __restrict__ src,
                            u16* __restrict__ dst, int n) {
  int i = blockIdx.x * blockDim.x + threadIdx.x;
  if (i < n) dst[i] = f2bf(src[i]);
}

// ---------------------------------------------------------------------------
// Kernel: row LayerNorm (f32 in) -> bf16 out.
// Row r maps to source row: (r/inner)*outer_stride + (r%inner) + first
// (handles the music_features[:, 2:, :] slice with inner=512, stride=514).
// ---------------------------------------------------------------------------
__global__ __launch_bounds__(128)
void ln_bf16_kernel(const float* __restrict__ src, u16* __restrict__ dst,
                    const float* __restrict__ g, const float* __restrict__ b,
                    int inner, int outer_stride_rows, int first) {
  int r = blockIdx.x;
  int tid = threadIdx.x;
  long srow = (long)(r / inner) * outer_stride_rows + (r % inner) + first;
  const float* x = src + srow * D;

  float v[4]; float s = 0.f, sq = 0.f;
#pragma unroll
  for (int i = 0; i < 4; ++i) {
    v[i] = x[tid + i * 128];
    s += v[i]; sq += v[i] * v[i];
  }
  __shared__ float2 red[128];
  red[tid] = make_float2(s, sq);
  __syncthreads();
  for (int o = 64; o > 0; o >>= 1) {
    if (tid < o) { red[tid].x += red[tid + o].x; red[tid].y += red[tid + o].y; }
    __syncthreads();
  }
  float mean = red[0].x * (1.f / D);
  float var  = red[0].y * (1.f / D) - mean * mean;
  float rstd = rsqrtf(var + 1e-5f);
#pragma unroll
  for (int i = 0; i < 4; ++i) {
    int c = tid + i * 128;
    dst[(size_t)r * D + c] = f2bf((v[i] - mean) * rstd * g[c] + b[c]);
  }
}

// ---------------------------------------------------------------------------
// Kernel: C[rows x 512] = A(bf16) @ W(bf16, row j = output col j)^T + bias.
// Block: 256 threads (8 waves). Block tile 128 rows x 64 cols.
// Each wave: 16 rows x 64 cols = 4 f32 accumulators, K in chunks of 32.
//   => 16 k-steps * 4 = 64 v_wmma_f32_16x16x32_bf16 per wave.
// W tile is staged TRANSPOSED in LDS (WT[k][n]) and each B fragment reads a
// WT column (stride 64 halves), giving the B layout
// (lane%16 = N, halves = K).
// ---------------------------------------------------------------------------
__global__ __launch_bounds__(256)
void gemm_bias_kernel(const u16* __restrict__ A, const u16* __restrict__ W,
                      const float* __restrict__ bias, void* __restrict__ C,
                      int c_is_f32) {
  __shared__ __align__(32) u16 As[128][32];
  __shared__ __align__(32) u16 WT[32][64];   // [k][n] transposed weight tile

  const int row0 = blockIdx.x * 128;
  const int col0 = blockIdx.y * 64;
  const int tid  = threadIdx.x;
  const int wave = tid >> 5;
  const int lane = tid & 31;

  const int m     = lane & 15;          // A-frag row within wave tile
  const int kbase = (lane >> 4) * 8;    // A-frag K sub-offset (lane group)

  v8f acc[4] = {v8f{}, v8f{}, v8f{}, v8f{}};

  for (int k0 = 0; k0 < D; k0 += 32) {
    __syncthreads();
    // Stage A tile: 128x32 bf16, 512 x 16B chunks, 2 per thread.
#pragma unroll
    for (int i = 0; i < 2; ++i) {
      int idx = i * 256 + tid;
      int r = idx >> 2, c = (idx & 3) * 8;
      *(uint4*)&As[r][c] =
          *(const uint4*)&A[(size_t)(row0 + r) * D + k0 + c];
    }
    // Stage W tile transposed: read W[j][k] rows coalesced, scatter to WT[k][j].
    {
      int r = tid >> 2, c = (tid & 3) * 8;   // r = local out-col, c = local k
      u16 tmp[8];
      *(uint4*)tmp = *(const uint4*)&W[(size_t)(col0 + r) * D + k0 + c];
#pragma unroll
      for (int j = 0; j < 8; ++j) WT[c + j][r] = tmp[j];
    }
    if (k0 + 32 < D)
      __builtin_prefetch(&A[(size_t)(row0 + (tid >> 1)) * D + k0 + 32], 0, 0);
    __syncthreads();

    // A fragment (16x32): lane=M, halves 0..7 -> K kbase+0..7, 8..15 -> +16.
    // Two contiguous 8-half runs -> 2x ds_load_b128.
    v16bf a;
#pragma unroll
    for (int h = 0; h < 16; ++h) {
      int k = kbase + (h & 7) + ((h >> 3) << 4);
      a[h] = us2bf(As[wave * 16 + m][k]);
    }
    // B fragments (32x16): lane%16=N, halves=K in the same order as A.
#pragma unroll
    for (int t = 0; t < 4; ++t) {
      v16bf bm;
#pragma unroll
      for (int s = 0; s < 16; ++s)
        bm[s] = us2bf(WT[kbase + (s & 7) + ((s >> 3) << 4)][t * 16 + m]);
      acc[t] = wmma_bf16(a, bm, acc[t]);
    }
  }

  // Epilogue: C layout lane<16 -> N=lane,M=r ; lane>=16 -> N=lane-16,M=r+8.
  const int n  = lane & 15;
  const int rb = (lane >> 4) * 8;
#pragma unroll
  for (int t = 0; t < 4; ++t) {
    float bv = bias[col0 + t * 16 + n];
#pragma unroll
    for (int r = 0; r < 8; ++r) {
      size_t grow = (size_t)(row0 + wave * 16 + rb + r);
      size_t gcol = (size_t)(col0 + t * 16 + n);
      float val = acc[t][r] + bv;
      if (c_is_f32) ((float*)C)[grow * D + gcol] = val;
      else          ((u16*)C)[grow * D + gcol]   = f2bf(val);
    }
  }
}

// ---------------------------------------------------------------------------
// Attention kernel: one wave per (16-video-row tile, m, h).
// Phase 1: logits(16x512) = Q(16x64) @ K^T via WMMA (K staged transposed).
// Phase 2: masked softmax over f (512), probs -> bf16 in LDS.
// Phase 3: attn(16x64) = P(16x512) @ V(512x64) via WMMA.
// ---------------------------------------------------------------------------
__global__ __launch_bounds__(32)
void attn_kernel(const u16* __restrict__ QB, const u16* __restrict__ KB,
                 const u16* __restrict__ VB, const int* __restrict__ mask,
                 u16* __restrict__ attnb) {
  __shared__ __align__(32) u16   Qs[16][64];
  __shared__ __align__(32) u16   KT[64][16];   // [hd][f] transposed chunk
  __shared__ __align__(32) u16   Vt[32][64];   // [f][hd] chunk
  __shared__ __align__(32) float Lg[16][512];  // logits / exp
  __shared__ __align__(32) u16   Pw[16][512];  // probs (bf16)

  const int v0   = blockIdx.x * 16;
  const int mcl  = blockIdx.y;
  const int h    = blockIdx.z;
  const int lane = threadIdx.x;

  const int mrow = lane & 15;
  const int kb   = (lane >> 4) * 8;

  // Load Q tile (16x64 bf16) coalesced.
#pragma unroll
  for (int i = 0; i < 4; ++i) {
    int idx = i * 32 + lane;
    int r = idx >> 3, c = (idx & 7) * 8;
    *(uint4*)&Qs[r][c] =
        *(const uint4*)&QB[(size_t)(v0 + r) * D + h * HD + c];
  }
  __syncthreads();

  // Pre-build both Q A-fragments (hd chunks 0..31 and 32..63).
  v16bf qf[2];
#pragma unroll
  for (int c2 = 0; c2 < 2; ++c2)
#pragma unroll
    for (int hh = 0; hh < 16; ++hh) {
      int k = c2 * 32 + kb + (hh & 7) + ((hh >> 3) << 4);
      qf[c2][hh] = us2bf(Qs[mrow][k]);
    }

  const float scale = 0.125f;  // 1/sqrt(64)

  // ---- Phase 1: logits --------------------------------------------------
  for (int f0 = 0; f0 < 512; f0 += 16) {
    // Stage K chunk transposed: KT[hd][f]; B-frags read down its columns (16-half stride).
#pragma unroll
    for (int i = 0; i < 4; ++i) {
      int idx = i * 32 + lane;
      int r = idx >> 3, c = (idx & 7) * 8;
      u16 tmp[8];
      *(uint4*)tmp =
          *(const uint4*)&KB[((size_t)mcl * 512 + f0 + r) * D + h * HD + c];
#pragma unroll
      for (int j = 0; j < 8; ++j) KT[c + j][r] = tmp[j];
    }
    __syncthreads();

    v8f acc = v8f{};
#pragma unroll
    for (int c2 = 0; c2 < 2; ++c2) {
      // B[k][n] = K[f0+n][c2*32 + k] = KT[c2*32+k][n], n = lane%16, k as in A
      v16bf bm;
#pragma unroll
      for (int s = 0; s < 16; ++s)
        bm[s] = us2bf(KT[c2 * 32 + kb + (s & 7) + ((s >> 3) << 4)][mrow]);
      acc = wmma_bf16(qf[c2], bm, acc);
    }

    const int n  = lane & 15;
    const int rb = (lane >> 4) * 8;
#pragma unroll
    for (int r = 0; r < 8; ++r) {
      float lg = acc[r] * scale;
      if (mask[mcl * 512 + f0 + n] == 0) lg = NEG_INF;
      Lg[rb + r][f0 + n] = lg;
    }
    __syncthreads();
  }

  // ---- Phase 2: softmax (lanes 0..15 own one row each) -------------------
  if (lane < 16) {
    float mx = NEG_INF;
    for (int f = 0; f < 512; ++f) mx = fmaxf(mx, Lg[lane][f]);
    float s = 0.f;
    for (int f = 0; f < 512; ++f) {
      float e = __expf(Lg[lane][f] - mx);
      s += e;
      Lg[lane][f] = e;
    }
    float inv = 1.f / s;
    for (int f = 0; f < 512; ++f) Pw[lane][f] = f2bf(Lg[lane][f] * inv);
  }
  __syncthreads();

  // ---- Phase 3: attn = P @ V --------------------------------------------
  v8f oacc[4] = {v8f{}, v8f{}, v8f{}, v8f{}};
  for (int fc = 0; fc < 512; fc += 32) {
    // Stage V chunk 32x64.
#pragma unroll
    for (int i = 0; i < 8; ++i) {
      int idx = i * 32 + lane;
      int r = idx >> 3, c = (idx & 7) * 8;
      *(uint4*)&Vt[r][c] =
          *(const uint4*)&VB[((size_t)mcl * 512 + fc + r) * D + h * HD + c];
    }
    __syncthreads();

    v16bf a;
#pragma unroll
    for (int hh = 0; hh < 16; ++hh) {
      int k = kb + (hh & 7) + ((hh >> 3) << 4);
      a[hh] = us2bf(Pw[mrow][fc + k]);
    }
#pragma unroll
    for (int t = 0; t < 4; ++t) {
      // B[k][n] = V[fc+k][t*16+n] = Vt[k][t*16 + n], n = lane%16, k as in A
      v16bf bm;
#pragma unroll
      for (int s = 0; s < 16; ++s)
        bm[s] = us2bf(Vt[kb + (s & 7) + ((s >> 3) << 4)][t * 16 + mrow]);
      oacc[t] = wmma_bf16(a, bm, oacc[t]);
    }
    __syncthreads();
  }

  const int n  = lane & 15;
  const int rb = (lane >> 4) * 8;
#pragma unroll
  for (int t = 0; t < 4; ++t)
#pragma unroll
    for (int r = 0; r < 8; ++r) {
      size_t vv = (size_t)(v0 + rb + r);
      attnb[(vv * 128 + mcl) * D + h * HD + t * 16 + n] = f2bf(oacc[t][r]);
    }
}

// ---------------------------------------------------------------------------
// Kernel: final residual + LayerNorm: out = LN(t + u, g3, b3), f32 out.
// ---------------------------------------------------------------------------
__global__ __launch_bounds__(128)
void final_ln_kernel(const u16* __restrict__ tb, const float* __restrict__ u,
                     const float* __restrict__ g, const float* __restrict__ b,
                     float* __restrict__ out) {
  int r = blockIdx.x;
  int tid = threadIdx.x;
  float v[4]; float s = 0.f, sq = 0.f;
#pragma unroll
  for (int i = 0; i < 4; ++i) {
    int c = tid + i * 128;
    float x = bf2f(tb[(size_t)r * D + c]) + u[(size_t)r * D + c];
    v[i] = x; s += x; sq += x * x;
  }
  __shared__ float2 red[128];
  red[tid] = make_float2(s, sq);
  __syncthreads();
  for (int o = 64; o > 0; o >>= 1) {
    if (tid < o) { red[tid].x += red[tid + o].x; red[tid].y += red[tid + o].y; }
    __syncthreads();
  }
  float mean = red[0].x * (1.f / D);
  float var  = red[0].y * (1.f / D) - mean * mean;
  float rstd = rsqrtf(var + 1e-5f);
#pragma unroll
  for (int i = 0; i < 4; ++i) {
    int c = tid + i * 128;
    out[(size_t)r * D + c] = (v[i] - mean) * rstd * g[c] + b[c];
  }
}

// ---------------------------------------------------------------------------
// Host-side orchestration
// ---------------------------------------------------------------------------
extern "C" void kernel_launch(void* const* d_in, const int* in_sizes, int n_in,
                              void* d_out, int out_size, void* d_ws,
                              size_t ws_size, hipStream_t stream) {
  (void)in_sizes; (void)n_in; (void)out_size; (void)ws_size;

  const float* video = (const float*)d_in[0];   // 128 x 512
  const float* music = (const float*)d_in[1];   // 128 x 514 x 512
  const int*   masks = (const int*)  d_in[2];   // 128 x 512
  const float* Wmat[5] = {(const float*)d_in[3], (const float*)d_in[4],
                          (const float*)d_in[5], (const float*)d_in[6],
                          (const float*)d_in[7]};            // Wq Wk Wv Wo Wl
  const float* bq = (const float*)d_in[8];
  const float* bk = (const float*)d_in[9];
  const float* bv = (const float*)d_in[10];
  const float* bo = (const float*)d_in[11];
  const float* bl = (const float*)d_in[12];
  const float* g1 = (const float*)d_in[13];
  const float* g2 = (const float*)d_in[14];
  const float* g3 = (const float*)d_in[15];
  const float* b1 = (const float*)d_in[16];
  const float* b2 = (const float*)d_in[17];
  const float* b3 = (const float*)d_in[18];

  // Workspace layout (bytes). mfln region is recycled for attn/t buffers,
  // K region recycled for u, once their producers/consumers are done.
  char* ws = (char*)d_ws;
  u16*   MFLN  = (u16*)  (ws + 0);              // 65536x512 bf16 (67.1 MB)
  u16*   ATTNB = (u16*)  (ws + 0);              // 16384x512 bf16 (reuse)
  float* ATTN2 = (float*)(ws + 16777216ull);    // 16384x512 f32
  u16*   TB    = (u16*)  (ws + 50331648ull);    // 16384x512 bf16
  u16*   KB    = (u16*)  (ws + 67108864ull);    // 65536x512 bf16
  float* UB    = (float*)(ws + 67108864ull);    // 16384x512 f32 (reuse K)
  u16*   VB    = (u16*)  (ws + 134217728ull);   // 65536x512 bf16
  u16*   WB    = (u16*)  (ws + 201326592ull);   // 5 x 512x512 bf16
  u16*   VELN  = (u16*)  (ws + 203948032ull);   // 128x512 bf16
  u16*   QB    = (u16*)  (ws + 204079104ull);   // 128x512 bf16

  // 1) Convert weights to bf16.
  for (int w = 0; w < 5; ++w)
    f2bf_kernel<<<1024, 256, 0, stream>>>(Wmat[w], WB + (size_t)w * D * D,
                                          D * D);

  // 2) LayerNorm inputs -> bf16.
  ln_bf16_kernel<<<65536, 128, 0, stream>>>(music, MFLN, g1, b1,
                                            /*inner=*/512, /*stride=*/514,
                                            /*first=*/2);
  ln_bf16_kernel<<<128, 128, 0, stream>>>(video, VELN, g1, b1,
                                          /*inner=*/128, /*stride=*/0,
                                          /*first=*/0);

  // 3) Projections: q, K, V (bf16 outputs).
  gemm_bias_kernel<<<dim3(1, 8),   256, 0, stream>>>(VELN, WB + 0 * D * D, bq,
                                                     QB, 0);
  gemm_bias_kernel<<<dim3(512, 8), 256, 0, stream>>>(MFLN, WB + 1 * D * D, bk,
                                                     KB, 0);
  gemm_bias_kernel<<<dim3(512, 8), 256, 0, stream>>>(MFLN, WB + 2 * D * D, bv,
                                                     VB, 0);

  // 4) Attention: grid (V/16, M, H), one wave per block.
  attn_kernel<<<dim3(8, 128, 8), 32, 0, stream>>>(QB, KB, VB, masks, ATTNB);

  // 5) Output projection (f32 out), LN2 -> bf16.
  gemm_bias_kernel<<<dim3(128, 8), 256, 0, stream>>>(ATTNB, WB + 3 * D * D, bo,
                                                     ATTN2, 1);
  ln_bf16_kernel<<<16384, 128, 0, stream>>>(ATTN2, TB, g2, b2,
                                            /*inner=*/16384, /*stride=*/0,
                                            /*first=*/0);

  // 6) Linear residual branch u = t @ Wl^T + bl (f32 out).
  gemm_bias_kernel<<<dim3(128, 8), 256, 0, stream>>>(TB, WB + 4 * D * D, bl,
                                                     UB, 1);

  // 7) out = LN(t + u, g3, b3).
  final_ln_kernel<<<16384, 128, 0, stream>>>(TB, UB, g3, b3, (float*)d_out);
}
